// Self_Attention_13666585936663
// MI455X (gfx1250) — matrix operand order repaired and checked by reading, not compiled, not run on hardware
//
#include <hip/hip_runtime.h>
#include <hip/hip_bf16.h>

typedef __bf16 bf16_t;
typedef __attribute__((ext_vector_type(16))) __bf16 v16bf;
typedef __attribute__((ext_vector_type(8)))  float  v8f;

#define B_ 2
#define L_ 2048
#define D_ 1024
#define H_ 16
#define DH_ 64

__device__ __forceinline__ v8f wmma_bf16(v16bf a, v16bf b, v8f c) {
  return __builtin_amdgcn_wmma_f32_16x16x32_bf16(
      /*neg_a=*/false, a, /*neg_b=*/false, b,
      /*c_mod=*/(short)0, c, /*reuse_a=*/false, /*reuse_b=*/false);
}

// A-fragment (16xK tile, 16-bit): lane lr holds row lr; lanes<16 take
// K = c0+{0..7, 16..23}, lanes>=16 take K = c0+{8..15, 24..31}.
__device__ __forceinline__ v16bf load_a_frag(const bf16_t* __restrict__ base,
                                             int ld, int c0) {
  const int lane = threadIdx.x & 31;
  const int hl = lane >> 4, lr = lane & 15;
  const bf16_t* p = base + (size_t)lr * ld + c0 + hl * 8;
  v16bf f;
  ((uint4*)&f)[0] = *(const uint4*)(p);
  ((uint4*)&f)[1] = *(const uint4*)(p + 16);
  return f;
}

// B-fragment (Kx16 tile, 16-bit): lane lr holds column lr; lanes<16 take
// K = c0+{0..7, 16..23}, lanes>=16 take K = c0+{8..15, 24..31}.
__device__ __forceinline__ v16bf load_b_frag(const bf16_t* __restrict__ base,
                                             int ld, int c0) {
  const int lane = threadIdx.x & 31;
  const int hl = lane >> 4, lr = lane & 15;
  const bf16_t* p = base + (size_t)lr * ld + c0 + hl * 8;
  v16bf f;
  ((uint4*)&f)[0] = *(const uint4*)(p);
  ((uint4*)&f)[1] = *(const uint4*)(p + 16);
  return f;
}

__global__ void cvt_f32_to_bf16_x4(const float4* __restrict__ src,
                                   bf16_t* __restrict__ dst, int n4) {
  int i = blockIdx.x * blockDim.x + threadIdx.x;
  if (i < n4) {
    float4 v = src[i];
    bf16_t t[4] = {(bf16_t)v.x, (bf16_t)v.y, (bf16_t)v.z, (bf16_t)v.w};
    *(uint2*)(dst + (size_t)i * 4) = *(const uint2*)t;
  }
}

// Y[m][n] = sum_k X[m][k] * W[n][k] + b[n]   (nn.Linear: x @ W^T + b)
// One wave -> 32(M) x 64(N) tile, software-pipelined k-loop (double-buffered
// fragments). mat 0/1 -> Q/K row-major bf16; mat 2 -> V transposed per head
// Vt[b][h][d][l].
__global__ __launch_bounds__(256) void qkv_proj_kernel(
    const bf16_t* __restrict__ xb,
    const bf16_t* __restrict__ Wqb, const bf16_t* __restrict__ Wkb,
    const bf16_t* __restrict__ Wvb,
    const float* __restrict__ bq, const float* __restrict__ bk,
    const float* __restrict__ bv,
    bf16_t* __restrict__ Qb, bf16_t* __restrict__ Kb,
    bf16_t* __restrict__ Vt) {
  const int wave = (blockIdx.x * blockDim.x + threadIdx.x) >> 5;
  const int mat = blockIdx.y;
  const int mTile = wave >> 4;   // 0..127  (B*L/32)
  const int nTile = wave & 15;   // 0..15   (D/64)
  const int m0 = mTile * 32, n0 = nTile * 64;

  const bf16_t* W = (mat == 0) ? Wqb : (mat == 1) ? Wkb : Wvb;
  const float* bias = (mat == 0) ? bq : (mat == 1) ? bk : bv;

  const int lane = threadIdx.x & 31;
  const int hl = lane >> 4, lr = lane & 15;

  const bf16_t* xrow0 = xb + (size_t)m0 * D_;
  const bf16_t* xrow1 = xb + (size_t)(m0 + 16) * D_;

  v8f acc[2][4] = {};

  // prologue: fragments for k0 = 0
  v16bf a0 = load_a_frag(xrow0, D_, 0);
  v16bf a1 = load_a_frag(xrow1, D_, 0);
  v16bf bb[4];
#pragma unroll
  for (int f = 0; f < 4; ++f)
    bb[f] = load_b_frag(W + (size_t)(n0 + f * 16) * D_, D_, 0);

  for (int k0 = 0; k0 < D_ - 32; k0 += 32) {
    const int kn = k0 + 32;
    // prefetch next iteration's fragments (overlaps the WMMAs below)
    v16bf na0 = load_a_frag(xrow0, D_, kn);
    v16bf na1 = load_a_frag(xrow1, D_, kn);
    v16bf nb[4];
#pragma unroll
    for (int f = 0; f < 4; ++f)
      nb[f] = load_b_frag(W + (size_t)(n0 + f * 16) * D_, D_, kn);

#pragma unroll
    for (int f = 0; f < 4; ++f) {
      acc[0][f] = wmma_bf16(a0, bb[f], acc[0][f]);
      acc[1][f] = wmma_bf16(a1, bb[f], acc[1][f]);
    }
    a0 = na0;
    a1 = na1;
#pragma unroll
    for (int f = 0; f < 4; ++f) bb[f] = nb[f];
  }
  // epilogue: last k-step
#pragma unroll
  for (int f = 0; f < 4; ++f) {
    acc[0][f] = wmma_bf16(a0, bb[f], acc[0][f]);
    acc[1][f] = wmma_bf16(a1, bb[f], acc[1][f]);
  }

#pragma unroll
  for (int f = 0; f < 4; ++f) {
    float bbias = bias[n0 + f * 16 + lr];
#pragma unroll
    for (int g = 0; g < 2; ++g)
#pragma unroll
      for (int r = 0; r < 8; ++r) acc[g][f][r] += bbias;
  }

  if (mat < 2) {
    bf16_t* out = (mat == 0) ? Qb : Kb;
#pragma unroll
    for (int g = 0; g < 2; ++g)
#pragma unroll
      for (int f = 0; f < 4; ++f)
#pragma unroll
        for (int r = 0; r < 8; ++r) {
          int m = m0 + g * 16 + hl * 8 + r;
          out[(size_t)m * D_ + n0 + f * 16 + lr] = (bf16_t)acc[g][f][r];
        }
  } else {
    const int bidx = m0 / L_;
    const int h = n0 / DH_;
#pragma unroll
    for (int g = 0; g < 2; ++g) {
      const int l0 = (m0 % L_) + g * 16;
#pragma unroll
      for (int f = 0; f < 4; ++f) {
        int dd = f * 16 + lr;
        bf16_t tmp[8];
#pragma unroll
        for (int r = 0; r < 8; ++r) tmp[r] = (bf16_t)acc[g][f][r];
        size_t off =
            ((size_t)(bidx * H_ + h) * DH_ + dd) * L_ + l0 + hl * 8;
        *(uint4*)(Vt + off) = *(const uint4*)tmp;  // 8 contiguous bf16
      }
    }
  }
}

// Flash attention: one wave per (b, h, 16-query tile). Computes S^T = K * Q^T
// so lane == query column; online softmax stats live per-lane; exp(S^T) in the
// C-fragment layout is exactly the A-fragment layout of P for O += P*V.
__global__ __launch_bounds__(256) void attn_kernel(
    const bf16_t* __restrict__ Qb, const bf16_t* __restrict__ Kb,
    const bf16_t* __restrict__ Vt, float* __restrict__ out) {
  const int wave = (blockIdx.x * blockDim.x + threadIdx.x) >> 5;
  const int bh = wave >> 7;   // 0..31  (B*H)
  const int qt = wave & 127;  // 0..127 (L/16)
  const int b = bh >> 4, h = bh & 15;
  const int q0 = qt * 16;

  const int lane = threadIdx.x & 31;
  const int hl = lane >> 4, lr = lane & 15;
  const int qi = q0 + lr;  // this lane's query index

  const bf16_t* Qbase = Qb + ((size_t)(b * L_ + q0)) * D_ + h * DH_;
  const bf16_t* Kbase = Kb + ((size_t)(b * L_)) * D_ + h * DH_;
  const bf16_t* Vbase = Vt + (size_t)(b * H_ + h) * DH_ * L_;

  // Q^T operand, reused across all key blocks (contraction d = 64 -> 2 frags)
  v16bf qf0 = load_b_frag(Qbase, D_, 0);
  v16bf qf1 = load_b_frag(Qbase, D_, 32);

  v8f o[4] = {};
  float m_i = -1e30f, l_i = 0.0f;

  for (int j0 = 0; j0 <= q0 + 15; j0 += 32) {
    // Issue all global loads for this key block up front: 4 K A-fragments
    // for the score WMMAs plus 4 V B-fragments whose latency is hidden
    // behind the softmax VALU work.
    const bf16_t* kj0 = Kbase + (size_t)j0 * D_;
    const bf16_t* kj1 = Kbase + (size_t)(j0 + 16) * D_;
    v16bf ka00 = load_a_frag(kj0, D_, 0);
    v16bf ka01 = load_a_frag(kj0, D_, 32);
    v16bf ka10 = load_a_frag(kj1, D_, 0);
    v16bf ka11 = load_a_frag(kj1, D_, 32);
    v16bf vb[4];
#pragma unroll
    for (int f = 0; f < 4; ++f)
      vb[f] = load_b_frag(Vbase + (size_t)(f * 16) * L_, L_, j0);

    // S^T tiles: s0 -> keys j0..j0+15, s1 -> keys j0+16..j0+31 (lane=query)
    v8f s0 = {}, s1 = {};
    s0 = wmma_bf16(ka00, qf0, s0);
    s0 = wmma_bf16(ka01, qf1, s0);
    s1 = wmma_bf16(ka10, qf0, s1);
    s1 = wmma_bf16(ka11, qf1, s1);

    // scale 1/sqrt(DH), causal mask, per-query max
    float p0[8], p1[8];
    float mx = -1e30f;
#pragma unroll
    for (int r = 0; r < 8; ++r) {
      int j = j0 + hl * 8 + r;  // key index of C-fragment row r in this lane
      float v0 = s0[r] * 0.125f;
      if (j > qi) v0 = -1e30f;
      float v1 = s1[r] * 0.125f;
      if (j + 16 > qi) v1 = -1e30f;
      p0[r] = v0;
      p1[r] = v1;
      mx = fmaxf(mx, fmaxf(v0, v1));
    }
    mx = fmaxf(mx, __shfl_xor(mx, 16, 32));  // combine key-halves per query
    float m_new = fmaxf(m_i, mx);
    float alpha = __expf(m_i - m_new);

    float rs = 0.0f;
#pragma unroll
    for (int r = 0; r < 8; ++r) {
      p0[r] = __expf(p0[r] - m_new);
      p1[r] = __expf(p1[r] - m_new);
      rs += p0[r] + p1[r];
    }
    rs += __shfl_xor(rs, 16, 32);
    l_i = l_i * alpha + rs;
    m_i = m_new;

    // P (16 queries x 32 keys) A-fragment: lanes<16 need K {0..7,16..23} of
    // row M=lr, lanes>=16 need K {8..15,24..31} -- exactly p0[],p1[] in lane.
    v16bf pa;
#pragma unroll
    for (int r = 0; r < 8; ++r) {
      pa[r] = (bf16_t)p0[r];
      pa[8 + r] = (bf16_t)p1[r];
    }

    // rescale O: row r of the C fragment is query (hl*8 + r); its alpha lives
    // in lane (hl ? 8+r : r) (lanes i and i+16 both hold query i's stats).
#pragma unroll
    for (int r = 0; r < 8; ++r) {
      float ar = __shfl(alpha, hl ? (8 + r) : r, 32);
      o[0][r] *= ar;
      o[1][r] *= ar;
      o[2][r] *= ar;
      o[3][r] *= ar;
    }

    // O += P * V  (V fragments were prefetched above)
#pragma unroll
    for (int f = 0; f < 4; ++f) o[f] = wmma_bf16(pa, vb[f], o[f]);
  }

  float inv = 1.0f / l_i;
#pragma unroll
  for (int r = 0; r < 8; ++r) {
    float ir = __shfl(inv, hl ? (8 + r) : r, 32);
    size_t row = (size_t)(b * L_ + q0 + hl * 8 + r);
#pragma unroll
    for (int f = 0; f < 4; ++f)
      out[row * D_ + h * DH_ + f * 16 + lr] = o[f][r] * ir;
  }
}

extern "C" void kernel_launch(void* const* d_in, const int* in_sizes, int n_in,
                              void* d_out, int out_size, void* d_ws,
                              size_t ws_size, hipStream_t stream) {
  const float* x = (const float*)d_in[0];
  // d_in[1] = atten_mask (strict upper triangular causal) -- applied analytically
  const float* Wq = (const float*)d_in[2];
  const float* bq = (const float*)d_in[3];
  const float* Wk = (const float*)d_in[4];
  const float* bk = (const float*)d_in[5];
  const float* Wv = (const float*)d_in[6];
  const float* bv = (const float*)d_in[7];
  float* out = (float*)d_out;

  const size_t BLD = (size_t)B_ * L_ * D_;  // 4,194,304
  const size_t DD = (size_t)D_ * D_;        // 1,048,576

  bf16_t* xb = (bf16_t*)d_ws;
  bf16_t* Wqb = xb + BLD;
  bf16_t* Wkb = Wqb + DD;
  bf16_t* Wvb = Wkb + DD;
  bf16_t* Qb = Wvb + DD;
  bf16_t* Kb = Qb + BLD;
  bf16_t* Vt = Kb + BLD;

  // 1) fp32 -> bf16 conversions (vectorized x4)
  {
    int n4 = (int)(BLD / 4);
    cvt_f32_to_bf16_x4<<<(n4 + 255) / 256, 256, 0, stream>>>((const float4*)x,
                                                             xb, n4);
    int w4 = (int)(DD / 4);
    cvt_f32_to_bf16_x4<<<(w4 + 255) / 256, 256, 0, stream>>>((const float4*)Wq,
                                                             Wqb, w4);
    cvt_f32_to_bf16_x4<<<(w4 + 255) / 256, 256, 0, stream>>>((const float4*)Wk,
                                                             Wkb, w4);
    cvt_f32_to_bf16_x4<<<(w4 + 255) / 256, 256, 0, stream>>>((const float4*)Wv,
                                                             Wvb, w4);
  }

  // 2) QKV projections: (B*L/32)*(D/64) = 2048 wave-tiles per matrix,
  //    8 waves per 256-thread block -> 256 blocks, y = matrix index
  qkv_proj_kernel<<<dim3(256, 3), 256, 0, stream>>>(xb, Wqb, Wkb, Wvb, bq, bk,
                                                    bv, Qb, Kb, Vt);

  // 3) fused causal attention: B*H*(L/16) = 4096 wave-tiles
  attn_kernel<<<dim3(512), 256, 0, stream>>>(Qb, Kb, Vt, out);
}
